// Net_31911607009950
// MI455X (gfx1250) — matrix operand order repaired and checked by reading, not compiled, not run on hardware
//
#include <hip/hip_runtime.h>
#include <hip/hip_bf16.h>
#include <math.h>
#include <stdint.h>

typedef __attribute__((ext_vector_type(16))) _Float16 v16h;
typedef __attribute__((ext_vector_type(8)))  _Float16 v8h;
typedef __attribute__((ext_vector_type(8)))  float    v8f;
typedef int v4i_t __attribute__((vector_size(16)));   // matches builtin param type

#define GAS __attribute__((address_space(1)))
#define LAS __attribute__((address_space(3)))

#define WMMA_F16(a,b,c) __builtin_amdgcn_wmma_f32_16x16x32_f16(false,(a),false,(b),(short)0,(c),false,false)

// ---- CDNA5 async HBM->LDS copy (ASYNCcnt path), with safe fallback ---------
__device__ __forceinline__ void cp16_async(const _Float16* g, _Float16* l) {
#if __has_builtin(__builtin_amdgcn_global_load_async_to_lds_b128)
  __builtin_amdgcn_global_load_async_to_lds_b128(
      (GAS v4i_t*)(GAS void*)(void*)g,
      (LAS v4i_t*)(LAS void*)(void*)l, 0, 0);
#else
  *(v8h*)l = *(const v8h*)g;
#endif
}
__device__ __forceinline__ void async_wait0() {
#if __has_builtin(__builtin_amdgcn_global_load_async_to_lds_b128)
# if __has_builtin(__builtin_amdgcn_s_wait_asynccnt)
  __builtin_amdgcn_s_wait_asynccnt(0);
# else
  asm volatile("s_wait_asynccnt 0x0" ::: "memory");
# endif
#endif
}

// ---- fragment loaders (layouts per CDNA5 ISA 7.12.2) -----------------------
__device__ __forceinline__ v16h ldA(const _Float16* rowp, int lanehi) {
  const v8h* q = (const v8h*)(rowp + (lanehi << 3));
  v8h lo = q[0];          // k0 .. k0+7
  v8h hi = q[2];          // k0+16 .. k0+23
  v16h r;
#pragma unroll
  for (int i = 0; i < 8; ++i) { r[i] = lo[i]; r[i + 8] = hi[i]; }
  return r;
}
__device__ __forceinline__ v16h ldB(const _Float16* rowp, int lanehi) {
  const v8h* q = (const v8h*)(rowp + (lanehi << 3));
  v8h lo = q[0];
  v8h hi = q[2];
  v16h r;
#pragma unroll
  for (int i = 0; i < 8; ++i) { r[i] = lo[i]; r[i + 8] = hi[i]; }
  return r;
}

// ---- K0: pad+convert adjacency fp32 -> f16 (done once; halves A traffic) ---
__global__ void cvt_pad_f16(const float* __restrict__ A, _Float16* __restrict__ A16,
                            int M, int NN, int NNp) {
  int c = blockIdx.x * 256 + threadIdx.x;
  int r = blockIdx.y;
  if (c < NNp)
    A16[(long)r * NNp + c] =
        (r < M && c < NN) ? (_Float16)A[(long)r * NN + c] : (_Float16)0.f;
}

// ---- K1: C[M,NC] = act(A[M,K] @ W[K,NC] + bias), compile-time K,NC ---------
template<int K, int NC>
__global__ __launch_bounds__(128)
void gemm_rw(const float* __restrict__ A, const float* __restrict__ W,
             const float* __restrict__ bias, float* __restrict__ C,
             int M, int act) {
  __shared__ __align__(16) _Float16 Alds[64 * K];
  __shared__ __align__(16) _Float16 Wt[NC * K];
  const int tid = threadIdx.x, lane = tid & 31, w = tid >> 5;
  const int lanen = lane & 15, lanehi = lane >> 4;
  const int row0 = blockIdx.x * 64;

#pragma unroll 4
  for (int idx = tid; idx < K * NC; idx += 128) {            // Wt[n][k]=W[k][n]
    int k = idx / NC, n = idx - k * NC;
    Wt[n * K + k] = (_Float16)W[idx];
  }
#pragma unroll 4
  for (int idx = tid; idx < 64 * K; idx += 128) {
    int r = idx / K, k = idx - r * K;
    int gr = row0 + r;
    Alds[idx] = (gr < M) ? (_Float16)A[(long)gr * K + k] : (_Float16)0.f;
  }
  __syncthreads();

  constexpr int NT = NC / 16, NK = K / 32;
  v8f acc[NT];
#pragma unroll
  for (int i = 0; i < NT; ++i) acc[i] = {};
  v16h af[NK];
#pragma unroll
  for (int kc = 0; kc < NK; ++kc)
    af[kc] = ldA(&Alds[(w * 16 + lanen) * K + kc * 32], lanehi);

#pragma unroll
  for (int nt = 0; nt < NT; ++nt) {
#pragma unroll
    for (int kc = 0; kc < NK; ++kc) {
      v16h b = ldB(&Wt[(nt * 16 + lanen) * K + kc * 32], lanehi);
      acc[nt] = WMMA_F16(af[kc], b, acc[nt]);
    }
  }

#pragma unroll
  for (int nt = 0; nt < NT; ++nt) {
    int col = nt * 16 + lanen;
    float bv = bias ? bias[col] : 0.f;
#pragma unroll
    for (int v = 0; v < 8; ++v) {
      int row = row0 + w * 16 + lanehi * 8 + v;
      if (row < M) {
        float x = acc[nt][v] + bv;
        if (act) x = fmaxf(x, 0.f);
        C[(long)row * NC + col] = x;
      }
    }
  }
}

// ---- K2a: fp32-adjacency fallback (used only if d_ws too small) ------------
template<int NC>
__global__ __launch_bounds__(128)
void gemm_bigA(const float* __restrict__ A, const float* __restrict__ B,
               float* __restrict__ C, int M, int NN, int act) {
  __shared__ __align__(16) _Float16 Alds[2][64 * 64];
  __shared__ __align__(16) _Float16 Bt[2][NC * 64];
  const int tid = threadIdx.x, lane = tid & 31, w = tid >> 5;
  const int lanen = lane & 15, lanehi = lane >> 4;
  const int row0 = blockIdx.x * 64;
  constexpr int NT = NC / 16;
  v8f acc[NT];
#pragma unroll
  for (int i = 0; i < NT; ++i) acc[i] = {};

  const int nkb = (NN + 63) >> 6;
  auto stage = [&](int kb, int sb) {
    const int k0 = kb * 64;
#pragma unroll 8
    for (int idx = tid; idx < 64 * 64; idx += 128) {
      int r = idx >> 6, k = idx & 63;
      int gr = row0 + r, gk = k0 + k;
      Alds[sb][idx] = (gr < M && gk < NN) ? (_Float16)A[(long)gr * NN + gk]
                                          : (_Float16)0.f;
    }
#pragma unroll
    for (int idx = tid; idx < NC * 64; idx += 128) {
      int k = idx / NC, n = idx - k * NC;
      int gk = k0 + k;
      Bt[sb][n * 64 + k] =
          (gk < NN) ? (_Float16)B[(long)gk * NC + n] : (_Float16)0.f;
    }
  };

  stage(0, 0);
  __syncthreads();
  for (int kb = 0; kb < nkb; ++kb) {
    const int sb = kb & 1;
    if (kb + 1 < nkb) stage(kb + 1, sb ^ 1);
    if (kb + 2 < nkb) {
      int r = tid >> 1, gk = (kb + 2) * 64 + (tid & 1) * 32;
      int gr = row0 + r;
      if (gr < M && gk < NN) __builtin_prefetch(&A[(long)gr * NN + gk], 0, 1);
    }
#pragma unroll
    for (int kk = 0; kk < 2; ++kk) {
      v16h af = ldA(&Alds[sb][(w * 16 + lanen) * 64 + kk * 32], lanehi);
#pragma unroll
      for (int nt = 0; nt < NT; ++nt) {
        v16h b = ldB(&Bt[sb][(nt * 16 + lanen) * 64 + kk * 32], lanehi);
        acc[nt] = WMMA_F16(af, b, acc[nt]);
      }
    }
    __syncthreads();
  }
#pragma unroll
  for (int nt = 0; nt < NT; ++nt) {
    int col = nt * 16 + lanen;
#pragma unroll
    for (int v = 0; v < 8; ++v) {
      int row = row0 + w * 16 + lanehi * 8 + v;
      if (row < M) {
        float x = acc[nt][v];
        if (act) x = fmaxf(x, 0.f);
        C[(long)row * NC + col] = x;
      }
    }
  }
}

// ---- K2b: f16-adjacency main path. A16 is padded to 64-multiples, so the A
// tile staging is guard-free and uses GLOBAL_LOAD_ASYNC_TO_LDS_B128 (ASYNCcnt)
// double-buffered against the WMMA chain. -----------------------------------
template<int NC>
__global__ __launch_bounds__(128)
void gemm_bigA16(const _Float16* __restrict__ A16, const float* __restrict__ B,
                 float* __restrict__ C, int M, int NN, int NNp, int act) {
  __shared__ __align__(16) _Float16 Alds[2][64 * 64];
  __shared__ __align__(16) _Float16 Bt[2][NC * 64];
  const int tid = threadIdx.x, lane = tid & 31, w = tid >> 5;
  const int lanen = lane & 15, lanehi = lane >> 4;
  const int row0 = blockIdx.x * 64;
  constexpr int NT = NC / 16;
  v8f acc[NT];
#pragma unroll
  for (int i = 0; i < NT; ++i) acc[i] = {};

  const int nkb = (NN + 63) >> 6;
  auto stage = [&](int kb, int sb) {
    const int k0 = kb * 64;
#pragma unroll
    for (int s = 0; s < 4; ++s) {                 // 128 thr * 4 * 16B = 8KB tile
      int e = (tid + s * 128) * 8;                // half index in 64x64 tile
      int r = e >> 6, k = e & 63;
      cp16_async(&A16[(long)(row0 + r) * NNp + k0 + k], &Alds[sb][e]);
    }
#pragma unroll
    for (int idx = tid; idx < NC * 64; idx += 128) {
      int k = idx / NC, n = idx - k * NC;
      int gk = k0 + k;
      Bt[sb][n * 64 + k] =
          (gk < NN) ? (_Float16)B[(long)gk * NC + n] : (_Float16)0.f;
    }
  };

  stage(0, 0);
  async_wait0();
  __syncthreads();
  for (int kb = 0; kb < nkb; ++kb) {
    const int sb = kb & 1;
    if (kb + 1 < nkb) stage(kb + 1, sb ^ 1);      // async fills other buffer
#pragma unroll
    for (int kk = 0; kk < 2; ++kk) {
      v16h af = ldA(&Alds[sb][(w * 16 + lanen) * 64 + kk * 32], lanehi);
#pragma unroll
      for (int nt = 0; nt < NT; ++nt) {
        v16h b = ldB(&Bt[sb][(nt * 16 + lanen) * 64 + kk * 32], lanehi);
        acc[nt] = WMMA_F16(af, b, acc[nt]);
      }
    }
    async_wait0();                                // staged chunk landed in LDS
    __syncthreads();
  }
#pragma unroll
  for (int nt = 0; nt < NT; ++nt) {
    int col = nt * 16 + lanen;
#pragma unroll
    for (int v = 0; v < 8; ++v) {
      int row = row0 + w * 16 + lanehi * 8 + v;
      if (row < M) {
        float x = acc[nt][v];
        if (act) x = fmaxf(x, 0.f);
        C[(long)row * NC + col] = x;
      }
    }
  }
}

// ---- K3: flash-style non-local block: Hout = H2 + softmax(TH @ PH^T) @ G ---
__global__ __launch_bounds__(128)
void flash_nl(const float* __restrict__ TH, const float* __restrict__ PH,
              const float* __restrict__ G, const float* __restrict__ H2,
              float* __restrict__ Hout, int NN) {
  __shared__ __align__(16) _Float16 phil[32 * 72];
  __shared__ __align__(16) _Float16 gT[64 * 40];
  __shared__ __align__(16) _Float16 plds[4][16 * 40];
  const int tid = threadIdx.x, lane = tid & 31, w = tid >> 5;
  const int lanen = lane & 15, lanehi = lane >> 4;
  const int r0w = blockIdx.x * 64 + w * 16;
  const int myrow = r0w + lanen;
  const bool rvalid = myrow < NN;

  v16h a0, a1;
  {
    const float* tp = TH + (long)myrow * 64;
    const int k0 = lanehi * 8;
#pragma unroll
    for (int i = 0; i < 8; ++i) {
      a0[i]     = rvalid ? (_Float16)tp[k0 + i]          : (_Float16)0.f;
      a0[i + 8] = rvalid ? (_Float16)tp[k0 + 16 + i]     : (_Float16)0.f;
      a1[i]     = rvalid ? (_Float16)tp[32 + k0 + i]     : (_Float16)0.f;
      a1[i + 8] = rvalid ? (_Float16)tp[32 + k0 + 16 + i]: (_Float16)0.f;
    }
  }

  v8f O[4];
#pragma unroll
  for (int ft = 0; ft < 4; ++ft) O[ft] = {};
  float m[8], l[8];
#pragma unroll
  for (int v = 0; v < 8; ++v) { m[v] = -3.0e38f; l[v] = 0.f; }

  const int nchunks = (NN + 31) >> 5;
  for (int jc = 0; jc < nchunks; ++jc) {
    const int j0 = jc * 32;
#pragma unroll 4
    for (int idx = tid; idx < 32 * 64; idx += 128) {
      int r = idx >> 6, c = idx & 63;
      int gr = j0 + r;
      phil[r * 72 + c] = (gr < NN) ? (_Float16)PH[(long)gr * 64 + c] : (_Float16)0.f;
      gT[c * 40 + r]   = (gr < NN) ? (_Float16)G[(long)gr * 64 + c]  : (_Float16)0.f;
    }
    __syncthreads();

    float sv[2][8];
#pragma unroll
    for (int t = 0; t < 2; ++t) {
      const _Float16* bp = &phil[(t * 16 + lanen) * 72];
      v16h b0 = ldB(bp, lanehi);
      v16h b1 = ldB(bp + 32, lanehi);
      v8f s = {};
      s = WMMA_F16(a0, b0, s);
      s = WMMA_F16(a1, b1, s);
      int col = j0 + t * 16 + lanen;
      bool cm = col < NN;
#pragma unroll
      for (int v = 0; v < 8; ++v) sv[t][v] = cm ? s[v] : -3.0e38f;
    }

    float scale[8];
#pragma unroll
    for (int v = 0; v < 8; ++v) {
      float x = fmaxf(sv[0][v], sv[1][v]);
      x = fmaxf(x, __shfl_xor(x, 1, 32));
      x = fmaxf(x, __shfl_xor(x, 2, 32));
      x = fmaxf(x, __shfl_xor(x, 4, 32));
      x = fmaxf(x, __shfl_xor(x, 8, 32));
      float mn = fmaxf(m[v], x);
      scale[v] = __expf(m[v] - mn);
      m[v] = mn;
      float p0 = __expf(sv[0][v] - mn);
      float p1 = __expf(sv[1][v] - mn);
      sv[0][v] = p0; sv[1][v] = p1;
      float rs = p0 + p1;
      rs += __shfl_xor(rs, 1, 32);
      rs += __shfl_xor(rs, 2, 32);
      rs += __shfl_xor(rs, 4, 32);
      rs += __shfl_xor(rs, 8, 32);
      l[v] = l[v] * scale[v] + rs;
    }
#pragma unroll
    for (int ft = 0; ft < 4; ++ft)
#pragma unroll
      for (int v = 0; v < 8; ++v) O[ft][v] *= scale[v];

#pragma unroll
    for (int t = 0; t < 2; ++t)
#pragma unroll
      for (int v = 0; v < 8; ++v)
        plds[w][(lanehi * 8 + v) * 40 + t * 16 + lanen] = (_Float16)sv[t][v];

    v16h ap = ldA(&plds[w][lanen * 40], lanehi);
#pragma unroll
    for (int ft = 0; ft < 4; ++ft) {
      v16h bg = ldB(&gT[(ft * 16 + lanen) * 40], lanehi);
      O[ft] = WMMA_F16(ap, bg, O[ft]);
    }
    __syncthreads();
  }

#pragma unroll
  for (int ft = 0; ft < 4; ++ft) {
    int col = ft * 16 + lanen;
#pragma unroll
    for (int v = 0; v < 8; ++v) {
      int row = r0w + lanehi * 8 + v;
      if (row < NN) {
        float inv = 1.f / l[v];
        Hout[(long)row * 64 + col] = H2[(long)row * 64 + col] + O[ft][v] * inv;
      }
    }
  }
}

// ---- K4: A_rec = sigmoid(H @ H^T), 64x64 output tiles ----------------------
__global__ __launch_bounds__(128)
void arec_hhT(const float* __restrict__ H, float* __restrict__ Out, int NN) {
  __shared__ __align__(16) _Float16 Hr[64 * 72];
  __shared__ __align__(16) _Float16 Hc[64 * 72];
  const int tid = threadIdx.x, lane = tid & 31, w = tid >> 5;
  const int lanen = lane & 15, lanehi = lane >> 4;
  const int rb0 = blockIdx.y * 64, cb0 = blockIdx.x * 64;
#pragma unroll 4
  for (int idx = tid; idx < 64 * 64; idx += 128) {
    int r = idx >> 6, c = idx & 63;
    int gr = rb0 + r, gc = cb0 + r;
    Hr[r * 72 + c] = (gr < NN) ? (_Float16)H[(long)gr * 64 + c] : (_Float16)0.f;
    Hc[r * 72 + c] = (gc < NN) ? (_Float16)H[(long)gc * 64 + c] : (_Float16)0.f;
  }
  __syncthreads();
  const _Float16* ar = &Hr[(w * 16 + lanen) * 72];
  v16h a0 = ldA(ar, lanehi);
  v16h a1 = ldA(ar + 32, lanehi);
#pragma unroll
  for (int ct = 0; ct < 4; ++ct) {
    const _Float16* br = &Hc[(ct * 16 + lanen) * 72];
    v16h b0 = ldB(br, lanehi);
    v16h b1 = ldB(br + 32, lanehi);
    v8f s = {};
    s = WMMA_F16(a0, b0, s);
    s = WMMA_F16(a1, b1, s);
    int col = cb0 + ct * 16 + lanen;
#pragma unroll
    for (int v = 0; v < 8; ++v) {
      int row = rb0 + w * 16 + lanehi * 8 + v;
      if (row < NN && col < NN)
        Out[(long)row * NN + col] = 1.f / (1.f + __expf(-s[v]));
    }
  }
}

// ---- K5: row-wise softmax over 16 logits -----------------------------------
__global__ void softmax16_rows(const float* __restrict__ X,
                               float* __restrict__ Out, int M) {
  int r = blockIdx.x * blockDim.x + threadIdx.x;
  if (r >= M) return;
  const float* p = X + (long)r * 16;
  float mx = -3.0e38f;
#pragma unroll
  for (int i = 0; i < 16; ++i) mx = fmaxf(mx, p[i]);
  float e[16], s = 0.f;
#pragma unroll
  for (int i = 0; i < 16; ++i) { e[i] = __expf(p[i] - mx); s += e[i]; }
  float inv = 1.f / s;
#pragma unroll
  for (int i = 0; i < 16; ++i) Out[(long)r * 16 + i] = e[i] * inv;
}

extern "C" void kernel_launch(void* const* d_in, const int* in_sizes, int n_in,
                              void* d_out, int out_size, void* d_ws, size_t ws_size,
                              hipStream_t stream) {
  const int M = 10000, NN = 10000;
  const int NNp = 10048;                       // 157*64, pad for guard-free tiles
  const float* X   = (const float*)d_in[0];
  const float* A   = (const float*)d_in[1];
  const float* W1  = (const float*)d_in[2];
  const float* W2  = (const float*)d_in[3];
  const float* Wt  = (const float*)d_in[4];
  const float* Wp  = (const float*)d_in[5];
  const float* Wg  = (const float*)d_in[6];
  const float* Wd1 = (const float*)d_in[7];
  const float* bd1 = (const float*)d_in[8];
  const float* Wd2 = (const float*)d_in[9];
  const float* bd2 = (const float*)d_in[10];
  const float* W3  = (const float*)d_in[11];
  const float* W4  = (const float*)d_in[12];
  float* out = (float*)d_out;

  float* ws   = (float*)d_ws;
  float* bufA = ws;                            // 10000*128
  float* bufB = bufA + (size_t)M * 128;        // 10000*128
  float* H2   = bufB + (size_t)M * 128;        // 10000*64
  float* th   = H2 + (size_t)M * 64;
  float* ph   = th + (size_t)M * 64;
  float* gg   = ph + (size_t)M * 64;
  float* Hn   = gg + (size_t)M * 64;
  const size_t base_bytes = (size_t)(5 * 64 + 2 * 128) * M * sizeof(float); // 23.04MB
  const size_t a16_bytes  = (size_t)NNp * NNp * sizeof(_Float16);           // ~202MB
  const bool f16path = ws_size >= base_bytes + a16_bytes;
  _Float16* A16 = (_Float16*)((char*)d_ws + base_bytes);

  dim3 blk(128);
  const int gb = (M + 63) / 64;

  if (f16path) {                               // one-time A fp32->f16 (padded)
    dim3 cg((NNp + 255) / 256, NNp);
    cvt_pad_f16<<<cg, 256, 0, stream>>>(A, A16, M, NN, NNp);
  }
#define AGG(NC, Bsrc, Cdst, act)                                               \
  do {                                                                         \
    if (f16path)                                                               \
      gemm_bigA16<NC><<<gb, blk, 0, stream>>>(A16, Bsrc, Cdst, M, NN, NNp, act);\
    else                                                                       \
      gemm_bigA<NC><<<gb, blk, 0, stream>>>(A, Bsrc, Cdst, M, NN, act);        \
  } while (0)

  // H1 = relu(A @ (X @ W1))
  gemm_rw<128,128><<<gb, blk, 0, stream>>>(X, W1, nullptr, bufA, M, 0);
  AGG(128, bufA, bufB, 1);
  // H2 = relu(A @ (H1 @ W2))
  gemm_rw<128,64> <<<gb, blk, 0, stream>>>(bufB, W2, nullptr, bufA, M, 0);
  AGG(64, bufA, H2, 1);
  // theta/phi/g
  gemm_rw<64,64>  <<<gb, blk, 0, stream>>>(H2, Wt, nullptr, th, M, 0);
  gemm_rw<64,64>  <<<gb, blk, 0, stream>>>(H2, Wp, nullptr, ph, M, 0);
  gemm_rw<64,64>  <<<gb, blk, 0, stream>>>(H2, Wg, nullptr, gg, M, 0);
  // Hn = H2 + softmax(theta phi^T) g   (streamed, no N x N intermediate)
  flash_nl<<<gb, blk, 0, stream>>>(th, ph, gg, H2, Hn, NN);
  // A_rec = sigmoid(Hn Hn^T) -> second output region
  dim3 g2(gb, gb);
  arec_hhT<<<g2, blk, 0, stream>>>(Hn, out + (size_t)M * 16, NN);
  // dense1/dense2 (bias + relu fused)
  gemm_rw<64,64>  <<<gb, blk, 0, stream>>>(Hn, Wd1, bd1, bufA, M, 1);
  gemm_rw<64,64>  <<<gb, blk, 0, stream>>>(bufA, Wd2, bd2, bufB, M, 1);
  // Att = relu(A @ (H @ W3))
  gemm_rw<64,32>  <<<gb, blk, 0, stream>>>(bufB, W3, nullptr, bufA, M, 0);
  AGG(32, bufA, th, 1);
  // logits = A @ (Att @ W4); softmax over axis=1 -> first output region
  gemm_rw<32,16>  <<<gb, blk, 0, stream>>>(th, W4, nullptr, bufA, M, 0);
  AGG(16, bufA, bufB, 0);
  softmax16_rows<<<(M + 255) / 256, 256, 0, stream>>>(bufB, out, M);
#undef AGG
}
